// MultiHeadAttention_33569464386363
// MI455X (gfx1250) — matrix operand order repaired and checked by reading, not compiled, not run on hardware
//
#include <hip/hip_runtime.h>
#include <hip/hip_bf16.h>

// ---------------------------------------------------------------------------
// MI455X (gfx1250) wave32 WMMA implementation.
// ctx = GAMMA * flashattn(q,k,v) + (1-GAMMA) * (visible_M @ Vproj)
// K tiles staged into LDS via the Tensor Data Mover (double buffered),
// V consumed from a transposed bf16 copy so PV B-fragments are contiguous.
// ---------------------------------------------------------------------------

#define NTOK  4096
#define HID   512
#define NHEAD 8
#define DKV   64
#define GAMMA 0.5f
#define LNEPS 1e-5f

typedef __bf16        v16bf __attribute__((ext_vector_type(16)));
typedef float         v8f   __attribute__((ext_vector_type(8)));
typedef unsigned int  v4u   __attribute__((ext_vector_type(4)));
typedef int           v8i   __attribute__((ext_vector_type(8)));
typedef int           v4i   __attribute__((ext_vector_type(4)));

#if defined(__gfx1250__) && __has_builtin(__builtin_amdgcn_tensor_load_to_lds) && \
    __has_builtin(__builtin_amdgcn_s_wait_tensorcnt)
#define USE_TDM 1
#else
#define USE_TDM 0
#endif

__device__ __forceinline__ v8f wmma_bf16(v16bf a, v16bf b, v8f c) {
    return __builtin_amdgcn_wmma_f32_16x16x32_bf16(
        /*neg_a=*/false, a, /*neg_b=*/false, b,
        /*c_mod=*/(short)0, c, /*reuse_a=*/false, /*reuse_b=*/false);
}

// ---- fragment loaders (contiguous k = 16*hi + e per lane half) -------------------------
__device__ __forceinline__ v16bf load_a_f32(const float* A, int lda, int row0, int k0) {
    const int lane = threadIdx.x & 31, l = lane & 15, hi = lane >> 4;
    const float* row = A + (size_t)(row0 + l) * lda + k0;
    v16bf f;
#pragma unroll
    for (int v = 0; v < 8; ++v)
#pragma unroll
        for (int p = 0; p < 2; ++p) {
            int k = (hi ? 16 : 0) + 2 * v + p;
            f[2 * v + p] = (__bf16)row[k];
        }
    return f;
}
__device__ __forceinline__ v16bf load_a_bf16(const __bf16* A, int lda, int row0, int k0) {
    const int lane = threadIdx.x & 31, l = lane & 15, hi = lane >> 4;
    const __bf16* row = A + (size_t)(row0 + l) * lda + k0;
    v16bf f;
#pragma unroll
    for (int v = 0; v < 8; ++v)
#pragma unroll
        for (int p = 0; p < 2; ++p) {
            int k = (hi ? 16 : 0) + 2 * v + p;
            f[2 * v + p] = row[k];
        }
    return f;
}
// B[k][n] = W[n][k] (W row-major): lane reads 16 contiguous floats.
__device__ __forceinline__ v16bf load_bT_f32(const float* W, int ldw, int n0, int k0) {
    const int lane = threadIdx.x & 31, l = lane & 15, hi = lane >> 4;
    const float* row = W + (size_t)(n0 + l) * ldw + k0 + (hi ? 16 : 0);
    v16bf f;
#pragma unroll
    for (int e = 0; e < 16; ++e) f[e] = (__bf16)row[e];
    return f;
}
// B row-major [K x N]: frag[e] = B[k0 + hi*16 + e][n0 + l]
__device__ __forceinline__ v16bf load_b_bf16(const __bf16* B, int ldb, int k0, int n0) {
    const int lane = threadIdx.x & 31, l = lane & 15, hi = lane >> 4;
    const __bf16* p = B + (size_t)(k0 + (hi ? 16 : 0)) * ldb + n0 + l;
    v16bf f;
#pragma unroll
    for (int e = 0; e < 16; ++e) f[e] = p[(size_t)e * ldb];
    return f;
}

// ---------------------------------------------------------------------------
// Kernel 1: C[NTOK][HID](bf16) = A(f32) @ W^T + bias.  Optional transposed
// secondary output Ct[HID][NTOK] (used for V so PV B-frags read contiguous).
// ---------------------------------------------------------------------------
__global__ __launch_bounds__(256) void proj_kernel(const float* __restrict__ A,
                                                   const float* __restrict__ W,
                                                   const float* __restrict__ bias,
                                                   __bf16* __restrict__ C,
                                                   __bf16* __restrict__ Ct) {
    const int w = threadIdx.x >> 5;
    const int m0 = blockIdx.x * 64 + (w >> 1) * 16;
    const int n0 = blockIdx.y * 64 + (w & 1) * 32;
    const int lane = threadIdx.x & 31, l = lane & 15, hi = lane >> 4;
    v8f c0 = {}, c1 = {};
    for (int k0 = 0; k0 < HID; k0 += 32) {
        v16bf a  = load_a_f32(A, HID, m0, k0);
        v16bf b0 = load_bT_f32(W, HID, n0, k0);
        v16bf b1 = load_bT_f32(W, HID, n0 + 16, k0);
        c0 = wmma_bf16(a, b0, c0);
        c1 = wmma_bf16(a, b1, c1);
    }
    const float bv0 = bias[n0 + l], bv1 = bias[n0 + 16 + l];
#pragma unroll
    for (int r = 0; r < 8; ++r) {
        const int row = m0 + r + (hi ? 8 : 0);
        const __bf16 o0 = (__bf16)(c0[r] + bv0);
        const __bf16 o1 = (__bf16)(c1[r] + bv1);
        C[(size_t)row * HID + n0 + l]      = o0;
        C[(size_t)row * HID + n0 + 16 + l] = o1;
        if (Ct) {
            Ct[(size_t)(n0 + l) * NTOK + row]      = o0;
            Ct[(size_t)(n0 + 16 + l) * NTOK + row] = o1;
        }
    }
}

// ---------------------------------------------------------------------------
// Kernel 2: mctx(f32) = (1-GAMMA) * visible_M(f32 [4096x4096]) @ Vproj(bf16).
// visible_M streamed once from HBM (64 MB @ 23.3 TB/s), L2-resident after.
// ---------------------------------------------------------------------------
__global__ __launch_bounds__(256) void mctx_kernel(const float* __restrict__ VM,
                                                   const __bf16* __restrict__ Vp,
                                                   float* __restrict__ mctx) {
    const int w = threadIdx.x >> 5;
    const int m0 = blockIdx.x * 64 + (w >> 1) * 16;
    const int n0 = blockIdx.y * 64 + (w & 1) * 32;
    const int lane = threadIdx.x & 31, l = lane & 15, hi = lane >> 4;
    v8f c0 = {}, c1 = {};
    const float* arow = VM + (size_t)(m0 + l) * NTOK;
    for (int k0 = 0; k0 < NTOK; k0 += 32) {
        __builtin_prefetch(arow + k0 + 128, 0, 1);   // global_prefetch ahead of stream
        v16bf a  = load_a_f32(VM, NTOK, m0, k0);
        v16bf b0 = load_b_bf16(Vp, HID, k0, n0);
        v16bf b1 = load_b_bf16(Vp, HID, k0, n0 + 16);
        c0 = wmma_bf16(a, b0, c0);
        c1 = wmma_bf16(a, b1, c1);
    }
#pragma unroll
    for (int r = 0; r < 8; ++r) {
        const int row = m0 + r + (hi ? 8 : 0);
        mctx[(size_t)row * HID + n0 + l]      = (1.0f - GAMMA) * c0[r];
        mctx[(size_t)row * HID + n0 + 16 + l] = (1.0f - GAMMA) * c1[r];
    }
}

// ---------------------------------------------------------------------------
// Kernel 3: flash attention per head. 128 threads = 4 waves x 16 query rows.
// K tiles (32x64 bf16) DMA'd into double-buffered LDS by the TDM (wave 0),
// overlapped with WMMA via s_wait_tensorcnt(1).  V read from the transposed
// copy (contiguous per-lane 32B).  P routed through LDS for C->A relayout.
// ---------------------------------------------------------------------------
#if USE_TDM
__device__ __forceinline__ void tdm_load_k(const __bf16* Kb, int j0, int hoff,
                                           unsigned lds_byte_off) {
    const unsigned long long ga =
        (unsigned long long)(size_t)(Kb + (size_t)j0 * HID + hoff);
    v4u g0;
    g0[0] = 1u;                                             // count=1 valid descriptor
    g0[1] = lds_byte_off;                                   // lds_addr
    g0[2] = (unsigned)ga;                                   // global_addr[31:0]
    g0[3] = (unsigned)((ga >> 32) & 0x01FFFFFFu) | 0x80000000u; // addr[56:32] | type=2
    v8i g1;
    g1[0] = 0x00010000;          // workgroup_mask=0, data_size=1 (2 bytes)
    g1[1] = (int)(512u << 16);   // tensor_dim0 = 512
    g1[2] = (int)(4096u << 16);  // tensor_dim1 = 4096
    g1[3] = (int)(64u << 16);    // tile_dim0 = 64 elements (128B rows)
    g1[4] = 32;                  // tile_dim1 = 32 rows
    g1[5] = 512;                 // tensor_dim0_stride = 512 elements
    g1[6] = 0;
    g1[7] = 0;
    const v4i z4 = {0, 0, 0, 0};               // groups 2/3 unused (2D tile)
    const v8i z8 = {0, 0, 0, 0, 0, 0, 0, 0};   // extra descriptor words (unused)
    __builtin_amdgcn_tensor_load_to_lds(g0, g1, z4, z4, z8, 0);
}
#endif

__device__ __forceinline__ void coop_load_k(__bf16* dst, const __bf16* Kb,
                                            int j0, int hoff) {
    for (int i = threadIdx.x; i < 256; i += 128) {           // 32 rows x 8 uint4
        const int r = i >> 3, q4 = i & 7;
        ((uint4*)dst)[i] = ((const uint4*)(Kb + (size_t)(j0 + r) * HID + hoff))[q4];
    }
}

__global__ __launch_bounds__(128) void attn_kernel(const __bf16* __restrict__ Qb,
                                                   const __bf16* __restrict__ Kb,
                                                   const __bf16* __restrict__ Vbt,
                                                   const float* __restrict__ mctx,
                                                   __bf16* __restrict__ ctxb) {
    const int h = blockIdx.y;
    const int w = threadIdx.x >> 5;
    const int lane = threadIdx.x & 31, l = lane & 15, hi = lane >> 4;
    const int m0 = blockIdx.x * 64 + w * 16;
    const int hoff = h * DKV;

    __shared__ __align__(16) __bf16 kt[2][32 * 64];          // double-buffered K tile
    __shared__ __bf16 pbuf[4][16][32];

    // q fragments (two K-halves of dk=64), pre-scaled by 1/sqrt(dk)
    v16bf qf[2];
#pragma unroll
    for (int ds = 0; ds < 2; ++ds) {
        const __bf16* qrow = Qb + (size_t)(m0 + l) * HID + hoff + ds * 32;
#pragma unroll
        for (int v = 0; v < 8; ++v)
#pragma unroll
            for (int p = 0; p < 2; ++p) {
                int k = (hi ? 16 : 0) + 2 * v + p;
                qf[ds][2 * v + p] = (__bf16)((float)qrow[k] * 0.125f);
            }
    }

    v8f ctx[4] = {};
    float rmax[8], rsum[8];
#pragma unroll
    for (int r = 0; r < 8; ++r) { rmax[r] = -3.0e38f; rsum[r] = 0.0f; }

    const int NIT = NTOK / 32;
    // preload buffer 0
#if USE_TDM
    if (w == 0) tdm_load_k(Kb, 0, hoff, (unsigned)(size_t)&kt[0][0]);
#else
    coop_load_k(&kt[0][0], Kb, 0, hoff);
#endif

    for (int it = 0; it < NIT; ++it) {
        const int j0  = it * 32;
        const int jn  = (it + 1 < NIT) ? j0 + 32 : j0;        // clamp final prefetch
        const int cur = it & 1, nxt = cur ^ 1;
        __syncthreads();                                       // kt[nxt] free to refill
#if USE_TDM
        if (w == 0) {
            tdm_load_k(Kb, jn, hoff, (unsigned)(size_t)&kt[nxt][0]);
            __builtin_amdgcn_s_wait_tensorcnt(1);              // kt[cur] DMA complete
        }
#else
        coop_load_k(&kt[nxt][0], Kb, jn, hoff);
#endif
        __syncthreads();                                       // kt[cur] visible to all
        const __bf16* kcur = &kt[cur][0];

        // S = q @ k^T : two 16x16 tiles (keys j0..+15 and j0+16..+31)
        v8f s0 = {}, s1 = {};
#pragma unroll
        for (int ds = 0; ds < 2; ++ds) {
            v16bf b0, b1;
            const __bf16* k0row = kcur + (size_t)l * 64 + ds * 32 + (hi ? 16 : 0);
            const __bf16* k1row = kcur + (size_t)(16 + l) * 64 + ds * 32 + (hi ? 16 : 0);
#pragma unroll
            for (int e = 0; e < 16; ++e) { b0[e] = k0row[e]; b1[e] = k1row[e]; }
            s0 = wmma_bf16(qf[ds], b0, s0);
            s1 = wmma_bf16(qf[ds], b1, s1);
        }

        // online softmax over the 32 new keys
#pragma unroll
        for (int r = 0; r < 8; ++r) {
            float mx = fmaxf(s0[r], s1[r]);
#pragma unroll
            for (int off = 1; off < 16; off <<= 1)
                mx = fmaxf(mx, __shfl_xor(mx, off, 32));       // within 16-lane half
            const float nm = fmaxf(rmax[r], mx);
            const float alpha = __expf(rmax[r] - nm);
            rmax[r] = nm;
#pragma unroll
            for (int dt = 0; dt < 4; ++dt) ctx[dt][r] *= alpha;
            const float p0 = __expf(s0[r] - nm);
            const float p1 = __expf(s1[r] - nm);
            float ls = p0 + p1;
#pragma unroll
            for (int off = 1; off < 16; off <<= 1)
                ls += __shfl_xor(ls, off, 32);
            rsum[r] = rsum[r] * alpha + ls;
            pbuf[w][r + (hi ? 8 : 0)][l]      = (__bf16)p0;    // C-layout -> LDS
            pbuf[w][r + (hi ? 8 : 0)][16 + l] = (__bf16)p1;
        }
        __syncthreads();   // order wave-local P stores before A-frag reload

        // reload P as a 16x32 A fragment
        v16bf pa;
#pragma unroll
        for (int v = 0; v < 8; ++v)
#pragma unroll
            for (int p = 0; p < 2; ++p) {
                int jj = (hi ? 16 : 0) + 2 * v + p;
                pa[2 * v + p] = pbuf[w][l][jj];
            }

        // ctx += P @ V : B-frags contiguous from transposed V copy (global/L2)
#pragma unroll
        for (int dt = 0; dt < 4; ++dt) {
            v16bf bv;
            const __bf16* vrow =
                Vbt + (size_t)(hoff + dt * 16 + l) * NTOK + j0 + (hi ? 16 : 0);
#pragma unroll
            for (int e = 0; e < 16; ++e) bv[e] = vrow[e];
            ctx[dt] = wmma_bf16(pa, bv, ctx[dt]);
        }
    }

    // epilogue: blend with precomputed (1-GAMMA)*visible_M@Vproj
#pragma unroll
    for (int r = 0; r < 8; ++r) {
        const int row = m0 + r + (hi ? 8 : 0);
        const float inv = GAMMA / rsum[r];
#pragma unroll
        for (int dt = 0; dt < 4; ++dt) {
            const int col = hoff + dt * 16 + l;
            const float val = ctx[dt][r] * inv + mctx[(size_t)row * HID + col];
            ctxb[(size_t)row * HID + col] = (__bf16)val;
        }
    }
}

// ---------------------------------------------------------------------------
// Kernel 4: X(f32) = ctx(bf16) @ Wo^T + bo + Vin  (residual fused)
// ---------------------------------------------------------------------------
__global__ __launch_bounds__(256) void outproj_kernel(const __bf16* __restrict__ Ctx,
                                                      const float* __restrict__ Wo,
                                                      const float* __restrict__ bo,
                                                      const float* __restrict__ Vin,
                                                      float* __restrict__ X) {
    const int w = threadIdx.x >> 5;
    const int m0 = blockIdx.x * 64 + (w >> 1) * 16;
    const int n0 = blockIdx.y * 64 + (w & 1) * 32;
    const int lane = threadIdx.x & 31, l = lane & 15, hi = lane >> 4;
    v8f c0 = {}, c1 = {};
    for (int k0 = 0; k0 < HID; k0 += 32) {
        v16bf a  = load_a_bf16(Ctx, HID, m0, k0);
        v16bf b0 = load_bT_f32(Wo, HID, n0, k0);
        v16bf b1 = load_bT_f32(Wo, HID, n0 + 16, k0);
        c0 = wmma_bf16(a, b0, c0);
        c1 = wmma_bf16(a, b1, c1);
    }
    const float bv0 = bo[n0 + l], bv1 = bo[n0 + 16 + l];
#pragma unroll
    for (int r = 0; r < 8; ++r) {
        const int row = m0 + r + (hi ? 8 : 0);
        X[(size_t)row * HID + n0 + l] =
            c0[r] + bv0 + Vin[(size_t)row * HID + n0 + l];
        X[(size_t)row * HID + n0 + 16 + l] =
            c1[r] + bv1 + Vin[(size_t)row * HID + n0 + 16 + l];
    }
}

// ---------------------------------------------------------------------------
// Kernel 5: row-wise layernorm. One wave per row (512 = 32 lanes x 16).
// ---------------------------------------------------------------------------
__global__ __launch_bounds__(256) void ln_kernel(const float* __restrict__ X,
                                                 const float* __restrict__ g,
                                                 const float* __restrict__ b,
                                                 float* __restrict__ out) {
    const int row = blockIdx.x * 8 + (threadIdx.x >> 5);
    const int lane = threadIdx.x & 31;
    const float* x = X + (size_t)row * HID;
    float vals[16];
    float s = 0.0f;
#pragma unroll
    for (int i = 0; i < 16; ++i) { vals[i] = x[lane + 32 * i]; s += vals[i]; }
#pragma unroll
    for (int off = 1; off < 32; off <<= 1) s += __shfl_xor(s, off, 32);
    const float mu = s * (1.0f / HID);
    float vs = 0.0f;
#pragma unroll
    for (int i = 0; i < 16; ++i) { const float d = vals[i] - mu; vs += d * d; }
#pragma unroll
    for (int off = 1; off < 32; off <<= 1) vs += __shfl_xor(vs, off, 32);
    const float rstd = rsqrtf(vs * (1.0f / HID) + LNEPS);
#pragma unroll
    for (int i = 0; i < 16; ++i) {
        const int c = lane + 32 * i;
        out[(size_t)row * HID + c] = (vals[i] - mu) * rstd * g[c] + b[c];
    }
}

// ---------------------------------------------------------------------------
extern "C" void kernel_launch(void* const* d_in, const int* in_sizes, int n_in,
                              void* d_out, int out_size, void* d_ws, size_t ws_size,
                              hipStream_t stream) {
    (void)in_sizes; (void)n_in; (void)out_size; (void)ws_size;
    const float* Q    = (const float*)d_in[0];
    const float* K    = (const float*)d_in[1];
    const float* V    = (const float*)d_in[2];
    // d_in[3] = attn_mask (unused by reference)
    const float* VM   = (const float*)d_in[4];
    const float* Wq   = (const float*)d_in[5];
    const float* bq   = (const float*)d_in[6];
    const float* Wk   = (const float*)d_in[7];
    const float* bk   = (const float*)d_in[8];
    const float* Wv   = (const float*)d_in[9];
    const float* bv   = (const float*)d_in[10];
    const float* Wo   = (const float*)d_in[11];
    const float* bo   = (const float*)d_in[12];
    const float* ln_g = (const float*)d_in[13];
    const float* ln_b = (const float*)d_in[14];
    float* outp = (float*)d_out;

    char* ws = (char*)d_ws;
    const size_t nbf = (size_t)NTOK * HID * sizeof(__bf16);   // 4 MB
    const size_t nf  = (size_t)NTOK * HID * sizeof(float);    // 8 MB
    __bf16* qb   = (__bf16*)ws;              ws += nbf;
    __bf16* kb   = (__bf16*)ws;              ws += nbf;
    __bf16* vb   = (__bf16*)ws;              ws += nbf;
    __bf16* vbt  = (__bf16*)ws;              ws += nbf;       // transposed Vproj
    __bf16* ctxb = (__bf16*)ws;              ws += nbf;
    float*  mctx = (float*)ws;               ws += nf;
    float*  xbuf = (float*)ws;               ws += nf;

    const dim3 gtile(NTOK / 64, HID / 64);   // (64, 8)
    proj_kernel<<<gtile, 256, 0, stream>>>(Q, Wq, bq, qb, (__bf16*)nullptr);
    proj_kernel<<<gtile, 256, 0, stream>>>(K, Wk, bk, kb, (__bf16*)nullptr);
    proj_kernel<<<gtile, 256, 0, stream>>>(V, Wv, bv, vb, vbt);
    mctx_kernel<<<gtile, 256, 0, stream>>>(VM, vb, mctx);
    attn_kernel<<<dim3(NTOK / 64, NHEAD), 128, 0, stream>>>(qb, kb, vbt, mctx, ctxb);
    outproj_kernel<<<gtile, 256, 0, stream>>>(ctxb, Wo, bo, V, xbuf);
    ln_kernel<<<NTOK / 8, 256, 0, stream>>>(xbuf, ln_g, ln_b, outp);
}
